// BilinearDiscriminator_70171175682175
// MI455X (gfx1250) — matrix operand order repaired and checked by reading, not compiled, not run on hardware
//
#include <hip/hip_runtime.h>
#include <hip/hip_fp16.h>
#include <hip/hip_bf16.h>

typedef __attribute__((ext_vector_type(16))) _Float16 v16h;
typedef __attribute__((ext_vector_type(8)))  _Float16 v8h;
typedef __attribute__((ext_vector_type(4)))  _Float16 v4h;
typedef __attribute__((ext_vector_type(8)))  float    v8f;

#define D_DIM 256
#define N_DIM 8192
#define M_DIM 8192

#define BK   32      // K-step per WMMA stage
#define LDT  40      // padded LDS row stride in halfs (80 B) -> conflict-free b128 frag loads

// CDNA5 async global->LDS copy (ASYNCcnt-tracked), if the toolchain exposes it.
// Probe-confirmed signature: (v4i addrspace(1)* gsrc, v4i addrspace(3)* lds, imm off, imm cpol)
#if __has_builtin(__builtin_amdgcn_global_load_async_to_lds_b128)
  #define HAVE_ASYNC_LDS 1
  typedef int v4i __attribute__((vector_size(16)));
  typedef __attribute__((address_space(1))) v4i gv4i;   // global (prints as __device__)
  typedef __attribute__((address_space(3))) v4i lv4i;   // LDS    (prints as __shared__)
#else
  #define HAVE_ASYNC_LDS 0
#endif

__device__ __forceinline__ void async_copy_b128(const void* gsrc, void* ldst) {
#if HAVE_ASYNC_LDS
  __builtin_amdgcn_global_load_async_to_lds_b128((gv4i*)gsrc, (lv4i*)ldst,
                                                 /*imm offset=*/0, /*cpol=*/0);
#else
  *(uint4*)ldst = *(const uint4*)gsrc;
#endif
}

__device__ __forceinline__ void wait_async_copies() {
#if HAVE_ASYNC_LDS
  #if __has_builtin(__builtin_amdgcn_s_wait_asynccnt)
    __builtin_amdgcn_s_wait_asynccnt(0);
  #else
    asm volatile("s_wait_asynccnt 0x0" ::: "memory");
  #endif
#endif
}

// ---------------------------------------------------------------------------
// Phase 1: XT[n][d] = sum_k x[n][k] * W[d][k]   (f32 accumulate, store f16)
// 1.07 GFLOP total -- negligible; plain FMA with x rows staged in LDS.
// ---------------------------------------------------------------------------
__global__ __launch_bounds__(256) void xt_gemm_kernel(const float* __restrict__ x,
                                                      const float* __restrict__ W,
                                                      _Float16* __restrict__ XT) {
  __shared__ float xs[16][D_DIM];
  const int row0 = blockIdx.x * 16;
  const int d0   = blockIdx.y * 16;
  const int tid  = threadIdx.x;

  for (int i = tid; i < 16 * D_DIM; i += 256) {
    int r = i >> 8, c = i & (D_DIM - 1);
    xs[r][c] = x[(row0 + r) * D_DIM + c];
  }
  __syncthreads();

  const int r  = tid >> 4;   // 0..15 row within tile
  const int dc = tid & 15;   // 0..15 output column within tile
  const float* wrow = W + (size_t)(d0 + dc) * D_DIM;   // W is L2-resident (256 KB)
  float acc = 0.f;
  #pragma unroll 8
  for (int k = 0; k < D_DIM; ++k) acc = fmaf(xs[r][k], wrow[k], acc);
  XT[(size_t)(row0 + r) * D_DIM + (d0 + dc)] = (_Float16)acc;
}

// ---------------------------------------------------------------------------
// Phase 2: Yh = f16(y)  (elementwise, vectorized x4)
// ---------------------------------------------------------------------------
__global__ __launch_bounds__(256) void y_cvt_kernel(const float* __restrict__ y,
                                                    _Float16* __restrict__ Yh) {
  int i = blockIdx.x * 256 + threadIdx.x;          // index of a float4 group
  const float4 v = ((const float4*)y)[i];
  v4h h;
  h[0] = (_Float16)v.x; h[1] = (_Float16)v.y;
  h[2] = (_Float16)v.z; h[3] = (_Float16)v.w;
  ((v4h*)Yh)[i] = h;
}

// ---------------------------------------------------------------------------
// Phase 3: out = sigmoid(XT @ Yh^T)
// 128x128 block tile, 8 waves; wave tile 32x64 = 2x4 WMMA tiles.
// Double-buffered LDS fed by async global->LDS DMA; v_wmma_f32_16x16x32_f16;
// fused sigmoid; non-temporal output stores (keep f16 operands in L2).
// ---------------------------------------------------------------------------
__global__ __launch_bounds__(256) void disc_gemm_kernel(const _Float16* __restrict__ XT,
                                                        const _Float16* __restrict__ Yh,
                                                        float* __restrict__ out) {
  __shared__ __align__(16) _Float16 As[2][128 * LDT];
  __shared__ __align__(16) _Float16 Bs[2][128 * LDT];

  const int tid  = threadIdx.x;
  const int wave = tid >> 5;
  const int lane = tid & 31;
  const int row0 = blockIdx.y * 128;   // N (XT rows)
  const int col0 = blockIdx.x * 128;   // M (Yh rows)

  const int wrow = (wave >> 1) * 32;   // 4 wave-rows
  const int wcol = (wave & 1) * 64;    // 2 wave-cols

  v8f acc[2][4];
  #pragma unroll
  for (int i = 0; i < 2; ++i)
    #pragma unroll
    for (int j = 0; j < 4; ++j) acc[i][j] = 0.0f;

  // global -> LDS: 8 KB per operand per k-step; 2x b128 per thread per operand
  auto issueTile = [&](int kt, int buf) {
    #pragma unroll
    for (int i = 0; i < 2; ++i) {
      const int idx = tid * 2 + i;        // 0..511
      const int rr  = idx >> 2;           // tile row 0..127
      const int kb  = idx & 3;            // 16-byte chunk within 64 B row
      async_copy_b128(XT + (size_t)(row0 + rr) * D_DIM + kt * BK + kb * 8,
                      &As[buf][rr * LDT + kb * 8]);
      async_copy_b128(Yh + (size_t)(col0 + rr) * D_DIM + kt * BK + kb * 8,
                      &Bs[buf][rr * LDT + kb * 8]);
    }
  };

  // A fragment (16x32 f16): lanes 0-15 get K[0..8)+K[16..24) of row M=lane,
  // lanes 16-31 get K[8..16)+K[24..32) (ISA 7.12.2 layout).
  auto loadA = [&](int buf, int mt) -> v16h {
    const int mrow = wrow + mt * 16 + (lane & 15);
    const int koff = (lane < 16) ? 0 : 8;
    const _Float16* p = &As[buf][mrow * LDT + koff];
    v8h lo = *(const v8h*)(p);
    v8h hi = *(const v8h*)(p + 16);
    return __builtin_shufflevector(lo, hi, 0,1,2,3,4,5,6,7,8,9,10,11,12,13,14,15);
  };

  // B fragment (32x16 f16): lane n<16 holds K[0..8)+K[16..24) of column n,
  // lanes 16-31 hold K[8..16)+K[24..32) of column n-16 (matches A layout).
  auto loadB = [&](int buf, int nt) -> v16h {
    const int ncol = wcol + nt * 16 + (lane & 15);
    const int koff = (lane < 16) ? 0 : 8;
    const _Float16* p = &Bs[buf][ncol * LDT + koff];
    v8h lo = *(const v8h*)(p);
    v8h hi = *(const v8h*)(p + 16);
    return __builtin_shufflevector(lo, hi, 0,1,2,3,4,5,6,7,8,9,10,11,12,13,14,15);
  };

  issueTile(0, 0);
  wait_async_copies();
  __syncthreads();

  const int KT = D_DIM / BK;   // 8
  for (int kt = 0; kt < KT; ++kt) {
    const int buf = kt & 1;
    if (kt + 1 < KT) issueTile(kt + 1, buf ^ 1);  // async prefetch into other buffer

    v16h af[2], bf[4];
    #pragma unroll
    for (int i = 0; i < 2; ++i) af[i] = loadA(buf, i);
    #pragma unroll
    for (int j = 0; j < 4; ++j) bf[j] = loadB(buf, j);

    #pragma unroll
    for (int i = 0; i < 2; ++i)
      #pragma unroll
      for (int j = 0; j < 4; ++j)
        acc[i][j] = __builtin_amdgcn_wmma_f32_16x16x32_f16(
            false, af[i], false, bf[j], (short)0, acc[i][j], false, false);

    wait_async_copies();   // prefetch DMA drained while WMMAs executed
    __syncthreads();
  }

  // Epilogue: C layout -> lane n = col (n&15); VGPR r -> row r (lanes 0-15) / r+8 (16-31)
  #pragma unroll
  for (int i = 0; i < 2; ++i) {
    #pragma unroll
    for (int j = 0; j < 4; ++j) {
      const int trow = row0 + wrow + i * 16 + ((lane >> 4) << 3);
      const int tcol = col0 + wcol + j * 16 + (lane & 15);
      #pragma unroll
      for (int r = 0; r < 8; ++r) {
        const float v = acc[i][j][r];
        const float s = 1.0f / (1.0f + __expf(-v));
        __builtin_nontemporal_store(s, &out[(size_t)(trow + r) * M_DIM + tcol]);
      }
    }
  }
}

// ---------------------------------------------------------------------------
extern "C" void kernel_launch(void* const* d_in, const int* in_sizes, int n_in,
                              void* d_out, int out_size, void* d_ws, size_t ws_size,
                              hipStream_t stream) {
  const float* x = (const float*)d_in[0];   // [N, D]
  const float* y = (const float*)d_in[1];   // [M, D]
  const float* W = (const float*)d_in[2];   // [D, D]
  float* out = (float*)d_out;               // [N, M]

  _Float16* XT = (_Float16*)d_ws;                   // 4 MB
  _Float16* Yh = XT + (size_t)N_DIM * D_DIM;        // 4 MB

  dim3 g1(N_DIM / 16, D_DIM / 16);
  xt_gemm_kernel<<<g1, 256, 0, stream>>>(x, W, XT);

  y_cvt_kernel<<<(M_DIM * D_DIM / 4) / 256, 256, 0, stream>>>(y, Yh);

  dim3 g3(M_DIM / 128, N_DIM / 128);
  disc_gemm_kernel<<<g3, 256, 0, stream>>>(XT, Yh, out);
}
